// GridRNN_31430570672396
// MI455X (gfx1250) — matrix operand order repaired and checked by reading, not compiled, not run on hardware
//
#include <hip/hip_runtime.h>
#include <math.h>

// ---------------------------------------------------------------------------
// GridRNN (2D tanh grid RNN), H=512, B=8, 48x48 grid, depth 3, for gfx1250.
//
// Strategy (MI455X / CDNA5):
//  * All matmuls on v_wmma_f32_16x16x32_bf16 (f32 accumulate, bf16 operands).
//  * src/trg converted to bf16 once; weights pre-packed once into WMMA
//    B-fragment order -> every GEMM k-step is pure b128 loads + wmma
//    (no conversion VALU, no branches in the hot loop).
//  * Software-pipelined k-loops: fragment k+1 is loaded before WMMA k issues,
//    overlapping global/LDS loads with the matrix pipe.
//  * Recurrent state kept as bf16 grids hx/hy; they feed the wavefront A tiles
//    AND the next depth's input GEMM without reconversion.
//  * Wavefront: 95 diagonal kernels per depth; each cell stages its 16x1024
//    state tile (batch 8 padded to 16, [hx_above | hy_left], zero at borders)
//    into LDS in A-fragment order (32 KB), then 4 waves x 32 WMMA k-steps.
// ---------------------------------------------------------------------------

constexpr int Hc = 512;   // hidden
constexpr int Bc = 8;     // batch
constexpr int Sc = 48;    // SRC == TRG
constexpr int Dc = 3;     // depth

typedef __attribute__((ext_vector_type(16))) __bf16 v16bf;
typedef __attribute__((ext_vector_type(8)))  float  v8f;

union Frag32 {
  uint4 u[2];
  v16bf v;
  unsigned short s[16];
};

__device__ __forceinline__ unsigned short f32_to_bf16_rne(float f) {
  unsigned int u = __builtin_bit_cast(unsigned int, f);
  u += 0x7FFFu + ((u >> 16) & 1u);           // round-to-nearest-even
  return (unsigned short)(u >> 16);
}

// A-fragment (16x32 bf16, ISA layout): lane<16 -> K {kb..kb+7, kb+16..kb+23},
// lanes 16..31 shifted by +8. Both runs are 16B-contiguous in a row-major row.
__device__ __forceinline__ v16bf frag_from_bf16row(const unsigned short* __restrict__ row,
                                                   int kbase) {
  Frag32 f;
  f.u[0] = *(const uint4*)(row + kbase);
  f.u[1] = *(const uint4*)(row + kbase + 16);
  return f.v;
}

// ---------------------------------------------------------------------------
// Convert an f32 array to bf16 (for src/trg sequences).
// ---------------------------------------------------------------------------
__global__ void convert_bf16_kernel(const float* __restrict__ in,
                                    unsigned short* __restrict__ out, int n) {
  int idx = blockIdx.x * blockDim.x + threadIdx.x;
  if (idx < n) out[idx] = f32_to_bf16_rne(in[idx]);
}

// ---------------------------------------------------------------------------
// Pack a row-major f32 weight (K x N) into bf16 WMMA B-fragment order:
//   out[((kt*(N/16)+nt)*32 + lane)*16 + e] =
//       bf16( W[kt*32 + (lane>=16)*8 + e + (e>=8)*8, nt*16 + lane%16] )
// ---------------------------------------------------------------------------
__global__ void pack_weight_kernel(const float* __restrict__ W,
                                   unsigned short* __restrict__ out,
                                   int K, int N) {
  int tid = blockIdx.x * blockDim.x + threadIdx.x;
  int frags = (K >> 5) * (N >> 4);
  if (tid >= frags * 32) return;
  int fi = tid >> 5, lane = tid & 31;
  int NT = N >> 4;
  int kt = fi / NT, nt = fi - kt * NT;
  int n  = (nt << 4) + (lane & 15);
  int kb = (kt << 5) + ((lane >> 4) << 3);
  __align__(16) unsigned short tmp[16];
#pragma unroll
  for (int e = 0; e < 16; ++e) tmp[e] = f32_to_bf16_rne(W[(size_t)(kb + e + ((e >> 3) << 3)) * N + n]);
  uint4* dst = (uint4*)(out + (size_t)tid * 16);
  dst[0] = *(const uint4*)&tmp[0];
  dst[1] = *(const uint4*)&tmp[8];
}

// ---------------------------------------------------------------------------
// Input projection GEMM: pout[r, n] = A[r, :] @ W(512x512) + bias[n]
//   A rows (bf16): depth 0 -> broadcast src_bf/trg_bf; depth>0 -> hx/hy grid.
// Block = 128 threads (4 waves); wave w owns a 16-col tile. grid (M/16, H/64).
// ---------------------------------------------------------------------------
__global__ void __launch_bounds__(128)
input_gemm_kernel(const unsigned short* __restrict__ seq,   // (B,S,H) bf16 (depth 0)
                  const unsigned short* __restrict__ prev,  // (B*S*S, H) bf16 (depth>0)
                  const unsigned short* __restrict__ Wp,    // packed Wih, K=512
                  const float* __restrict__ bias,           // (H)
                  float* __restrict__ pout,                 // (B*S*S, H) f32
                  int use_seq, int seq_is_src) {
  const int lane   = threadIdx.x & 31;
  const int w      = threadIdx.x >> 5;
  const int rbase  = blockIdx.x << 4;
  const int ntile  = (blockIdx.y << 2) + w;       // 0..31
  const int m      = lane & 15;
  const int hiHalf = lane >> 4;                   // 0 | 1
  const int r      = rbase + m;

  // Resolve the A-row pointer ONCE (both paths bf16 => branch-free k-loop).
  const unsigned short* row;
  if (use_seq) {
    int b   = r / (Sc * Sc);
    int rem = r - b * Sc * Sc;
    int idx = seq_is_src ? (rem / Sc) : (rem - (rem / Sc) * Sc);   // i or j
    row = seq + ((size_t)b * Sc + idx) * Hc;
  } else {
    row = prev + (size_t)r * Hc;
  }
  const v16bf* Wfrag = (const v16bf*)Wp + (size_t)ntile * 32 + lane;  // + kt*32

  constexpr int KT = Hc / 32;                     // 16 k-steps
  const int kofs = hiHalf << 3;

  v8f c = {};
  // Software pipeline: prefetch fragment kt+1 before issuing WMMA kt.
  v16bf a_nxt = frag_from_bf16row(row, kofs);
  v16bf b_nxt = Wfrag[0];
#pragma unroll
  for (int kt = 0; kt < KT; ++kt) {
    v16bf a_cur = a_nxt;
    v16bf b_cur = b_nxt;
    if (kt + 1 < KT) {
      a_nxt = frag_from_bf16row(row, ((kt + 1) << 5) + kofs);
      b_nxt = Wfrag[(size_t)(kt + 1) * ((Hc >> 4) * 32) / 32];  // stride 32*32 v16bf / 32
    }
    c = __builtin_amdgcn_wmma_f32_16x16x32_bf16(false, a_cur, false, b_cur,
                                                (short)0, c, false, false);
  }

  const int n  = (ntile << 4) + (lane & 15);
  const float bn = bias[n];
#pragma unroll
  for (int v = 0; v < 8; ++v) {
    int mm = v + (hiHalf << 3);                   // C/D layout: lanes>=16 -> M+8
    pout[(size_t)(rbase + mm) * Hc + n] = c[v] + bn;
  }
}

// ---------------------------------------------------------------------------
// One wavefront diagonal step for one channel (ch=0: hx/Whx, ch=1: hy/Why):
//   cell (i,j), i+j=t:
//     val = tanh(pxy[b,i,j,:] + [hx(i-1,j) | hy(i,j-1)] @ Whh + bias)
// Block = 128 threads (4 waves); grid (ncells, H/64).
// ---------------------------------------------------------------------------
__global__ void __launch_bounds__(128)
diag_step_kernel(const unsigned short* __restrict__ hx_g,   // (B*S*S, H) bf16
                 const unsigned short* __restrict__ hy_g,   // (B*S*S, H) bf16
                 unsigned short* __restrict__ state_out,    // hx_g or hy_g
                 const unsigned short* __restrict__ Whp,    // packed Whh, K=1024
                 const float* __restrict__ pxy,             // px or py, f32
                 const float* __restrict__ bias,            // (H)
                 float* __restrict__ out,                   // (B,D,S,S,2,H) f32
                 int t, int d, int ch) {
  __shared__ __align__(32) unsigned short sA[16 * 1024];    // 32 KB state tile

  const int i_lo = (t > Sc - 1) ? (t - (Sc - 1)) : 0;
  const int i = i_lo + blockIdx.x;
  const int j = t - i;
  const int tid = threadIdx.x;

  // Stage A (16 rows x 1024 K) into LDS in fragment order: 1024 fragments of
  // 16 bf16; each thread fills 8 fragments with two 16B loads / two 16B stores.
#pragma unroll
  for (int q = 0; q < 8; ++q) {
    int fidx = tid * 8 + q;                   // 0..1023
    int kt = fidx >> 5;
    int lf = fidx & 31;
    int m  = lf & 15;
    int kb = (kt << 5) + ((lf >> 4) << 3);    // runs [kb,kb+8) and [kb+16,kb+24)
    uint4 lo = make_uint4(0u, 0u, 0u, 0u);
    uint4 hi = make_uint4(0u, 0u, 0u, 0u);
    if (m < Bc) {                             // rows 8..15 stay zero (M padding)
      int b = m;
      if (kb < Hc) {                          // first half of state: hx(i-1, j)
        if (i > 0) {
          const unsigned short* row = hx_g + (((size_t)b * Sc + (i - 1)) * Sc + j) * Hc;
          lo = *(const uint4*)(row + kb);
          hi = *(const uint4*)(row + kb + 16);
        }
      } else {                                // second half: hy(i, j-1)
        if (j > 0) {
          const unsigned short* row = hy_g + (((size_t)b * Sc + i) * Sc + (j - 1)) * Hc;
          lo = *(const uint4*)(row + (kb - Hc));
          hi = *(const uint4*)(row + (kb - Hc) + 16);
        }
      }
    }
    *(uint4*)&sA[(size_t)fidx * 16]     = lo;
    *(uint4*)&sA[(size_t)fidx * 16 + 8] = hi;
  }
  __syncthreads();

  const int lane   = tid & 31;
  const int w      = tid >> 5;
  const int ntile  = (blockIdx.y << 2) + w;   // 0..31
  const int hiHalf = lane >> 4;

  const v16bf* Afrag = (const v16bf*)sA + lane;                       // + kt*32
  const v16bf* Bfrag = (const v16bf*)Whp + (size_t)ntile * 32 + lane; // + kt*32*32

  constexpr int KT = 32;                      // K = 1024
  v8f c = {};
  // Software pipeline: prefetch LDS-A / global-B for kt+1 before WMMA kt.
  v16bf a_nxt = Afrag[0];
  v16bf b_nxt = Bfrag[0];
#pragma unroll 4
  for (int kt = 0; kt < KT; ++kt) {
    v16bf a_cur = a_nxt;
    v16bf b_cur = b_nxt;
    if (kt + 1 < KT) {
      a_nxt = Afrag[(size_t)(kt + 1) * 32];
      b_nxt = Bfrag[(size_t)(kt + 1) * 32 * 32];
    }
    c = __builtin_amdgcn_wmma_f32_16x16x32_bf16(false, a_cur, false, b_cur,
                                                (short)0, c, false, false);
  }

  const int n  = (ntile << 4) + (lane & 15);
  const float bn = bias[n];
#pragma unroll
  for (int v = 0; v < 8; ++v) {
    int m = v + (hiHalf << 3);
    if (m < Bc) {
      int b = m;
      size_t cell = ((size_t)b * Sc + i) * Sc + j;
      float val = tanhf(c[v] + pxy[cell * Hc + n] + bn);
      out[((((((size_t)b * Dc + d) * Sc + i) * Sc + j) * 2 + ch) * Hc) + n] = val;
      state_out[cell * Hc + n] = f32_to_bf16_rne(val);
    }
  }
}

// ---------------------------------------------------------------------------
extern "C" void kernel_launch(void* const* d_in, const int* in_sizes, int n_in,
                              void* d_out, int out_size, void* d_ws, size_t ws_size,
                              hipStream_t stream) {
  const float* src   = (const float*)d_in[0];   // (B,S,H)
  const float* trg   = (const float*)d_in[1];   // (B,S,H)
  const float* Wx_ih = (const float*)d_in[2];   // (D,H,H)
  const float* Wx_hh = (const float*)d_in[3];   // (D,2H,H)
  const float* bx_ih = (const float*)d_in[4];   // (D,H)
  const float* bx_hh = (const float*)d_in[5];   // (D,H)
  const float* Wy_ih = (const float*)d_in[6];   // (D,H,H)
  const float* Wy_hh = (const float*)d_in[7];   // (D,2H,H)
  const float* by_ih = (const float*)d_in[8];   // (D,H)
  const float* by_hh = (const float*)d_in[9];   // (D,H)
  // d_in[10] = depth scalar (device mem); reference fixes DEPTH=3, hardcoded.
  float* out = (float*)d_out;
  (void)in_sizes; (void)n_in; (void)out_size; (void)ws_size;

  // Workspace (~124 MB): packed bf16 weights, bf16 seqs, f32 px/py, bf16 hx/hy.
  char* ws = (char*)d_ws;
  size_t off = 0;
  auto carve = [&](size_t bytes) -> char* {
    char* p = ws + off;
    off += (bytes + 255) & ~(size_t)255;
    return p;
  };
  unsigned short* WxihP  = (unsigned short*)carve((size_t)Dc * Hc * Hc * 2);
  unsigned short* WyihP  = (unsigned short*)carve((size_t)Dc * Hc * Hc * 2);
  unsigned short* WxhhP  = (unsigned short*)carve((size_t)Dc * 2 * Hc * Hc * 2);
  unsigned short* WyhhP  = (unsigned short*)carve((size_t)Dc * 2 * Hc * Hc * 2);
  unsigned short* src_bf = (unsigned short*)carve((size_t)Bc * Sc * Hc * 2);
  unsigned short* trg_bf = (unsigned short*)carve((size_t)Bc * Sc * Hc * 2);
  float* px = (float*)carve((size_t)Bc * Sc * Sc * Hc * 4);
  float* py = (float*)carve((size_t)Bc * Sc * Sc * Hc * 4);
  unsigned short* hx = (unsigned short*)carve((size_t)Bc * Sc * Sc * Hc * 2);
  unsigned short* hy = (unsigned short*)carve((size_t)Bc * Sc * Sc * Hc * 2);

  // ---- one-time prologue: convert sequences, pack weights ----
  {
    int nseq = Bc * Sc * Hc;                                           // 196608
    convert_bf16_kernel<<<(nseq + 255) / 256, 256, 0, stream>>>(src, src_bf, nseq);
    convert_bf16_kernel<<<(nseq + 255) / 256, 256, 0, stream>>>(trg, trg_bf, nseq);
  }
  const int blocks_ih = ((Hc / 32) * (Hc / 16) * 32 + 255) / 256;      // 64
  const int blocks_hh = ((2 * Hc / 32) * (Hc / 16) * 32 + 255) / 256;  // 128
  for (int d = 0; d < Dc; ++d) {
    pack_weight_kernel<<<blocks_ih, 256, 0, stream>>>(
        Wx_ih + (size_t)d * Hc * Hc, WxihP + (size_t)d * Hc * Hc, Hc, Hc);
    pack_weight_kernel<<<blocks_ih, 256, 0, stream>>>(
        Wy_ih + (size_t)d * Hc * Hc, WyihP + (size_t)d * Hc * Hc, Hc, Hc);
    pack_weight_kernel<<<blocks_hh, 256, 0, stream>>>(
        Wx_hh + (size_t)d * 2 * Hc * Hc, WxhhP + (size_t)d * 2 * Hc * Hc, 2 * Hc, Hc);
    pack_weight_kernel<<<blocks_hh, 256, 0, stream>>>(
        Wy_hh + (size_t)d * 2 * Hc * Hc, WyhhP + (size_t)d * 2 * Hc * Hc, 2 * Hc, Hc);
  }

  const int Mrows = Bc * Sc * Sc;               // 18432
  dim3 ggrid(Mrows / 16, Hc / 64);              // (1152, 8)
  for (int d = 0; d < Dc; ++d) {
    // px/py projections (depth 0 reads broadcast src/trg; later depths hx/hy).
    input_gemm_kernel<<<ggrid, 128, 0, stream>>>(
        src_bf, hx, WxihP + (size_t)d * Hc * Hc, bx_ih + (size_t)d * Hc, px, d == 0, 1);
    input_gemm_kernel<<<ggrid, 128, 0, stream>>>(
        trg_bf, hy, WyihP + (size_t)d * Hc * Hc, by_ih + (size_t)d * Hc, py, d == 0, 0);

    // Wavefront over 95 anti-diagonals; both channels of diagonal t read only
    // diagonal t-1 state, so the two launches per t are order-independent.
    for (int t = 0; t < 2 * Sc - 1; ++t) {
      int ncells = Sc - ((t >= Sc) ? (t - (Sc - 1)) : ((Sc - 1) - t));
      dim3 dgrid(ncells, Hc / 64);
      diag_step_kernel<<<dgrid, 128, 0, stream>>>(
          hx, hy, hx, WxhhP + (size_t)d * 2 * Hc * Hc, px,
          bx_hh + (size_t)d * Hc, out, t, d, 0);
      diag_step_kernel<<<dgrid, 128, 0, stream>>>(
          hx, hy, hy, WyhhP + (size_t)d * 2 * Hc * Hc, py,
          by_hh + (size_t)d * Hc, out, t, d, 1);
    }
  }
}
